// GNN_82377472737433
// MI455X (gfx1250) — matrix operand order repaired and checked by reading, not compiled, not run on hardware
//
#include <hip/hip_runtime.h>

typedef __attribute__((ext_vector_type(16))) __bf16 v16bf;
typedef __attribute__((ext_vector_type(8)))  float  v8f;

#define FIN 128
#define FH  256

// ---------------- CSR build: counts ----------------
__global__ void k_zero(int* __restrict__ counts, int n) {
    int i = blockIdx.x * blockDim.x + threadIdx.x;
    if (i < n) counts[i] = 0;
}

__global__ void k_count(const int* __restrict__ dst, int* __restrict__ counts, int e) {
    int i = blockIdx.x * blockDim.x + threadIdx.x;
    if (i < e) atomicAdd(&counts[dst[i]], 1);
}

// single-block exclusive scan over counts -> offsets/cursor, plus dinv = rsqrt(indeg+1)
__global__ void k_scan(const int* __restrict__ counts, int* __restrict__ offsets,
                       int* __restrict__ cursor, float* __restrict__ dinv, int n) {
    __shared__ int warpsum[32];
    __shared__ int carry_s;
    if (threadIdx.x == 0) carry_s = 0;
    __syncthreads();
    const int lane = threadIdx.x & 31;
    const int wid  = threadIdx.x >> 5;
    for (int base = 0; base < n; base += (int)blockDim.x) {
        int i = base + (int)threadIdx.x;
        int v = (i < n) ? counts[i] : 0;
        int s = v;
#pragma unroll
        for (int o = 1; o < 32; o <<= 1) {
            int t = __shfl_up(s, o, 32);
            if (lane >= o) s += t;
        }
        if (lane == 31) warpsum[wid] = s;
        __syncthreads();
        if (wid == 0) {
            int w = warpsum[lane];
#pragma unroll
            for (int o = 1; o < 32; o <<= 1) {
                int t = __shfl_up(w, o, 32);
                if (lane >= o) w += t;
            }
            warpsum[lane] = w;
        }
        __syncthreads();
        int incl = s + (wid > 0 ? warpsum[wid - 1] : 0);
        int excl = incl - v + carry_s;
        if (i < n) {
            offsets[i] = excl;
            cursor[i]  = excl;
            dinv[i]    = rsqrtf((float)(v + 1));
        }
        __syncthreads();
        if (threadIdx.x == blockDim.x - 1) carry_s += incl;
        __syncthreads();
    }
    if (threadIdx.x == 0) offsets[n] = carry_s;
}

__global__ void k_scatter(const int* __restrict__ src, const int* __restrict__ dst,
                          int* __restrict__ cursor, int* __restrict__ csrc, int e) {
    int i = blockIdx.x * blockDim.x + threadIdx.x;
    if (i < e) {
        int pos = atomicAdd(&cursor[dst[i]], 1);
        csrc[pos] = src[i];
    }
}

// ---------------- W prep: split f32 W[K,N] into bf16 hi/lo, swizzled to WMMA B layout --
// one wave per (tn, kb) tile; lane's 16 K-elements stored contiguously (32B) so the
// GEMM fetches them as clause-able global_load_b128 pairs with zero conversion VALU.
__global__ void k_prep_w(const float* __restrict__ W, __bf16* __restrict__ Whi,
                         __bf16* __restrict__ Wlo, int K, int N) {
    const int lane = threadIdx.x & 31;
    const int tile = (blockIdx.x * blockDim.x + threadIdx.x) >> 5;
    const int KB = K >> 5;
    if (tile >= (N >> 4) * KB) return;
    const int tn  = tile / KB;
    const int kb  = tile - tn * KB;
    const int col = tn * 16 + (lane & 15);
    const int klo = (lane & 16) ? 16 : 0;
    const size_t base = ((size_t)tile * 32 + lane) * 16;
#pragma unroll 4
    for (int j = 0; j < 16; ++j) {
        float v = W[(size_t)(kb * 32 + ((j & 8) << 1) + (klo >> 1) + (j & 7)) * N + col];
        __bf16 h = (__bf16)v;
        Whi[base + j] = h;
        Wlo[base + j] = (__bf16)(v - (float)h);
    }
}

// ---------------- WMMA GEMM: C[M,N] = A[M,K] @ B_prepped, split-bf16 x3 ----------------
// Wave computes a 16x64 strip (4 accumulators); A split once per k-step, B from prepped
// layout (contiguous b128 loads). Block = 8 waves = 2 M-tiles x full N=256.
__global__ void k_gemm_wmma(const float* __restrict__ A, const __bf16* __restrict__ Bhi,
                            const __bf16* __restrict__ Blo, float* __restrict__ C,
                            int M, int K, int N) {
    const int lane = threadIdx.x & 31;
    const int wave = threadIdx.x >> 5;                 // 0..7
    const int tm   = blockIdx.x * 2 + (wave >> 2);     // M tile
    const int tng  = (wave & 3) * 4;                   // first of 4 N tiles
    if (tm * 16 >= M) return;                          // wave-uniform: EXEC stays all-ones

    const int KB   = K >> 5;
    const int rowA = tm * 16 + (lane & 15);
    const int ah   = (lane & 16) ? 8 : 0;              // A: lanes 16-31 hold K 8..15 / 24..31

    v8f acc[4] = {v8f{}, v8f{}, v8f{}, v8f{}};
    for (int kb = 0; kb < KB; ++kb) {
        // ---- A tile 16x32, split hi/lo bf16 (once, reused by 4 N-tiles) ----
        const float* ap = A + (size_t)rowA * K + kb * 32;
        v16bf a_hi, a_lo;
#pragma unroll
        for (int j = 0; j < 8; ++j) {
            float v0 = ap[ah + j];
            float v1 = ap[16 + ah + j];
            __bf16 h0 = (__bf16)v0, h1 = (__bf16)v1;
            a_hi[j]     = h0;  a_lo[j]     = (__bf16)(v0 - (float)h0);
            a_hi[8 + j] = h1;  a_lo[8 + j] = (__bf16)(v1 - (float)h1);
        }
#pragma unroll
        for (int t = 0; t < 4; ++t) {
            const size_t tb = (((size_t)(tng + t) * KB + kb) * 32 + lane);
            v16bf b_hi = ((const v16bf*)Bhi)[tb];
            v16bf b_lo = ((const v16bf*)Blo)[tb];
            acc[t] = __builtin_amdgcn_wmma_f32_16x16x32_bf16(false, a_hi, false, b_hi,
                                                             (short)0, acc[t], false, false);
            acc[t] = __builtin_amdgcn_wmma_f32_16x16x32_bf16(false, a_lo, false, b_hi,
                                                             (short)0, acc[t], false, false);
            acc[t] = __builtin_amdgcn_wmma_f32_16x16x32_bf16(false, a_hi, false, b_lo,
                                                             (short)0, acc[t], false, false);
        }
    }
    const int rowC = tm * 16 + ((lane & 16) ? 8 : 0);
#pragma unroll
    for (int t = 0; t < 4; ++t) {
        float* cp = C + (size_t)rowC * N + (tng + t) * 16 + (lane & 15);
#pragma unroll
        for (int r = 0; r < 8; ++r) cp[(size_t)r * N] = acc[t][r];
    }
}

// ---------------- fused aggregation (layer 1): gather-style, no atomics ----------------
__global__ void k_agg1(const float* __restrict__ xw, float* __restrict__ hout,
                       const float* __restrict__ dinv, const int* __restrict__ offs,
                       const int* __restrict__ csrc, const float* __restrict__ bias, int n) {
    const int lane = threadIdx.x & 31;
    const int node = (blockIdx.x * blockDim.x + threadIdx.x) >> 5;
    if (node >= n) return;
    const float di = dinv[node];
    const float4* pself = (const float4*)(xw + (size_t)node * FH) + lane * 2;
    float4 a0 = pself[0], a1 = pself[1];
    const float cs = di * di;
    float4 acc0 = make_float4(a0.x * cs, a0.y * cs, a0.z * cs, a0.w * cs);
    float4 acc1 = make_float4(a1.x * cs, a1.y * cs, a1.z * cs, a1.w * cs);
    const int beg = offs[node], end = offs[node + 1];
    for (int j = beg; j < end; ++j) {
        const int s = csrc[j];
        if (j + 1 < end) {
            __builtin_prefetch(xw + (size_t)csrc[j + 1] * FH + lane * 8, 0, 1);
        }
        const float cf = dinv[s] * di;
        const float4* ps = (const float4*)(xw + (size_t)s * FH) + lane * 2;
        float4 v0 = ps[0], v1 = ps[1];
        acc0.x += v0.x * cf;  acc0.y += v0.y * cf;  acc0.z += v0.z * cf;  acc0.w += v0.w * cf;
        acc1.x += v1.x * cf;  acc1.y += v1.y * cf;  acc1.z += v1.z * cf;  acc1.w += v1.w * cf;
    }
    const float4 b0 = ((const float4*)bias)[lane * 2];
    const float4 b1 = ((const float4*)bias)[lane * 2 + 1];
    acc0.x = fmaxf(acc0.x + b0.x, 0.0f);  acc0.y = fmaxf(acc0.y + b0.y, 0.0f);
    acc0.z = fmaxf(acc0.z + b0.z, 0.0f);  acc0.w = fmaxf(acc0.w + b0.w, 0.0f);
    acc1.x = fmaxf(acc1.x + b1.x, 0.0f);  acc1.y = fmaxf(acc1.y + b1.y, 0.0f);
    acc1.z = fmaxf(acc1.z + b1.z, 0.0f);  acc1.w = fmaxf(acc1.w + b1.w, 0.0f);
    float4* po = (float4*)(hout + (size_t)node * FH) + lane * 2;
    po[0] = acc0;  po[1] = acc1;
}

// ---------------- fused aggregation (layer 2) + readout GEMV ----------------
__global__ void k_agg2(const float* __restrict__ xw, const float* __restrict__ dinv,
                       const int* __restrict__ offs, const int* __restrict__ csrc,
                       const float* __restrict__ bias, const float* __restrict__ wfc,
                       const float* __restrict__ bfc, float* __restrict__ out, int n) {
    const int lane = threadIdx.x & 31;
    const int node = (blockIdx.x * blockDim.x + threadIdx.x) >> 5;
    if (node >= n) return;
    const float di = dinv[node];
    const float4* pself = (const float4*)(xw + (size_t)node * FH) + lane * 2;
    float4 a0 = pself[0], a1 = pself[1];
    const float cs = di * di;
    float4 acc0 = make_float4(a0.x * cs, a0.y * cs, a0.z * cs, a0.w * cs);
    float4 acc1 = make_float4(a1.x * cs, a1.y * cs, a1.z * cs, a1.w * cs);
    const int beg = offs[node], end = offs[node + 1];
    for (int j = beg; j < end; ++j) {
        const int s = csrc[j];
        if (j + 1 < end) {
            __builtin_prefetch(xw + (size_t)csrc[j + 1] * FH + lane * 8, 0, 1);
        }
        const float cf = dinv[s] * di;
        const float4* ps = (const float4*)(xw + (size_t)s * FH) + lane * 2;
        float4 v0 = ps[0], v1 = ps[1];
        acc0.x += v0.x * cf;  acc0.y += v0.y * cf;  acc0.z += v0.z * cf;  acc0.w += v0.w * cf;
        acc1.x += v1.x * cf;  acc1.y += v1.y * cf;  acc1.z += v1.z * cf;  acc1.w += v1.w * cf;
    }
    const float4 b0 = ((const float4*)bias)[lane * 2];
    const float4 b1 = ((const float4*)bias)[lane * 2 + 1];
    const float4 w0 = ((const float4*)wfc)[lane * 2];
    const float4 w1 = ((const float4*)wfc)[lane * 2 + 1];
    float s = fmaxf(acc0.x + b0.x, 0.0f) * w0.x + fmaxf(acc0.y + b0.y, 0.0f) * w0.y
            + fmaxf(acc0.z + b0.z, 0.0f) * w0.z + fmaxf(acc0.w + b0.w, 0.0f) * w0.w
            + fmaxf(acc1.x + b1.x, 0.0f) * w1.x + fmaxf(acc1.y + b1.y, 0.0f) * w1.y
            + fmaxf(acc1.z + b1.z, 0.0f) * w1.z + fmaxf(acc1.w + b1.w, 0.0f) * w1.w;
#pragma unroll
    for (int o = 16; o > 0; o >>= 1) s += __shfl_down(s, o, 32);
    if (lane == 0) out[node] = s + bfc[0];
}

extern "C" void kernel_launch(void* const* d_in, const int* in_sizes, int n_in,
                              void* d_out, int out_size, void* d_ws, size_t ws_size,
                              hipStream_t stream) {
    const float* x   = (const float*)d_in[0];
    const int*   ei  = (const int*)d_in[1];
    const float* W1  = (const float*)d_in[2];
    const float* b1  = (const float*)d_in[3];
    const float* W2  = (const float*)d_in[4];
    const float* b2  = (const float*)d_in[5];
    const float* Wfc = (const float*)d_in[6];
    const float* bfc = (const float*)d_in[7];
    float* out = (float*)d_out;

    const int n = in_sizes[0] / FIN;     // 50000
    const int e = in_sizes[1] / 2;       // 800000
    const int* src = ei;
    const int* dst = ei + e;

    // ---- workspace layout (byte offsets; total ~108 MB) ----
    char* ws = (char*)d_ws;
    float*  dinv    = (float*)(ws + 0x000000);     // n f32
    int*    counts  = (int*)  (ws + 0x040000);     // n i32
    int*    offsets = (int*)  (ws + 0x080000);     // n+1 i32
    int*    cursor  = (int*)  (ws + 0x0C0000);     // n i32
    int*    csrc    = (int*)  (ws + 0x100000);     // e i32 (3.2MB)
    __bf16* Whi1    = (__bf16*)(ws + 0x440000);    // 128x256 bf16 (64KB)
    __bf16* Wlo1    = (__bf16*)(ws + 0x460000);
    __bf16* Whi2    = (__bf16*)(ws + 0x480000);    // 256x256 bf16 (128KB)
    __bf16* Wlo2    = (__bf16*)(ws + 0x4C0000);
    float*  bufA    = (float*)(ws + 0x500000);     // n*FH f32 (X@W scratch)
    float*  bufB    = bufA + (size_t)n * FH;       // n*FH f32 (h1)

    const int T = 256;
    const int gemm_blocks = (n / 16 + 1) / 2;               // 2 M-tiles per block
    const int ptiles1 = (FH / 16) * (FIN / 32);             // 64 prep waves
    const int ptiles2 = (FH / 16) * (FH / 32);              // 128 prep waves

    // ---- CSR by destination (shared by both layers) + dinv ----
    k_zero<<<(n + T - 1) / T, T, 0, stream>>>(counts, n);
    k_count<<<(e + T - 1) / T, T, 0, stream>>>(dst, counts, e);
    k_scan<<<1, 1024, 0, stream>>>(counts, offsets, cursor, dinv, n);
    k_scatter<<<(e + T - 1) / T, T, 0, stream>>>(src, dst, cursor, csrc, e);

    // ---- weight prep (split-bf16, WMMA-layout swizzle) ----
    k_prep_w<<<(ptiles1 * 32 + T - 1) / T, T, 0, stream>>>(W1, Whi1, Wlo1, FIN, FH);
    k_prep_w<<<(ptiles2 * 32 + T - 1) / T, T, 0, stream>>>(W2, Whi2, Wlo2, FH, FH);

    // ---- layer 1 ----
    k_gemm_wmma<<<gemm_blocks, T, 0, stream>>>(x, Whi1, Wlo1, bufA, n, FIN, FH);
    k_agg1<<<(n + 7) / 8, T, 0, stream>>>(bufA, bufB, dinv, offsets, csrc, b1, n);

    // ---- layer 2 (+ fused readout) ----
    k_gemm_wmma<<<gemm_blocks, T, 0, stream>>>(bufB, Whi2, Wlo2, bufA, n, FH, FH);
    k_agg2<<<(n + 7) / 8, T, 0, stream>>>(bufA, dinv, offsets, csrc, b2, Wfc, bfc, out, n);
}
